// YOLOGNNWrapper_76914274337053
// MI455X (gfx1250) — matrix operand order repaired and checked by reading, not compiled, not run on hardware
//
#include <hip/hip_runtime.h>
#include <hip/hip_bf16.h>
#include <math.h>

// ---------------- problem constants ----------------
constexpr int BI    = 16;      // images
constexpr int NN    = 128;     // detections / image
constexpr int DR    = 12544;   // roi feature dim (divisible by 32)
constexpr int HID   = 128;
constexpr int PARTS = 22;
constexpr int ROWS  = BI * NN; // 2048
constexpr float DTH2 = 50.0f * 50.0f;
constexpr float CTH  = 0.8f;

// ---------------- vector types for WMMA ----------------
typedef __attribute__((ext_vector_type(16))) __bf16 v16bf;
typedef __attribute__((ext_vector_type(8)))  __bf16 bf16x8;
typedef __attribute__((ext_vector_type(8)))  float  v8f;
typedef __attribute__((ext_vector_type(2)))  float  v2f;

static __device__ inline v8f zero8() {
    v8f z;
#pragma unroll
    for (int i = 0; i < 8; ++i) z[i] = 0.0f;
    return z;
}

// =====================================================================
// Kernel 0a: fv f32 -> bf16 copy + per-row reciprocal L2 norm
// one block (256 thr) per row of fv
// =====================================================================
__global__ void convert_fv_kernel(const float* __restrict__ fv,
                                  __bf16* __restrict__ fvb,
                                  float* __restrict__ rn) {
    const int row = blockIdx.x;
    const int tid = threadIdx.x;
    const float* p = fv  + (size_t)row * DR;
    __bf16*      q = fvb + (size_t)row * DR;
    float ss = 0.0f;
    for (int k = tid; k < DR; k += 256) {
        float v = p[k];
        ss += v * v;
        q[k] = (__bf16)v;
    }
    __shared__ float red[256];
    red[tid] = ss;
    __syncthreads();
    for (int s = 128; s > 0; s >>= 1) {
        if (tid < s) red[tid] += red[tid + s];
        __syncthreads();
    }
    if (tid == 0) rn[row] = 1.0f / fmaxf(sqrtf(red[0]), 1e-12f);
}

// =====================================================================
// Kernel 0b: W1 [12546,128] f32 -> transposed bf16 W1t [128][12544]
// (last 2 input rows = ctr contribution handled in f32 later)
// =====================================================================
__global__ void convert_w1t_kernel(const float* __restrict__ W1,
                                   __bf16* __restrict__ w1t) {
    const int total = HID * DR;
    for (int idx = blockIdx.x * blockDim.x + threadIdx.x; idx < total;
         idx += gridDim.x * blockDim.x) {
        int h = idx / DR;
        int k = idx - h * DR;
        w1t[idx] = (__bf16)W1[(size_t)k * HID + h];   // write contiguous
    }
}

// =====================================================================
// Kernel 1: per-image Gram (bf16 WMMA, K=12544) + adjacency build
// one wave per 16x16 C tile; 16 images * 64 tiles = 1024 waves
// =====================================================================
__global__ void gram_adj_kernel(const __bf16* __restrict__ fvb,
                                const float*  __restrict__ rn,
                                const float*  __restrict__ ctr,
                                float* __restrict__ Araw) {
    const int wave = (blockIdx.x * blockDim.x + threadIdx.x) >> 5;
    const int lane = threadIdx.x & 31;
    const int b  = wave >> 6;
    const int t  = wave & 63;
    const int ti = t >> 3, tj = t & 7;
    const int la = lane & 15, lh = lane >> 4;

    const __bf16* base = fvb + (size_t)b * NN * DR;
    const __bf16* pa = base + (size_t)(ti * 16 + la) * DR;   // A row (M)
    const __bf16* pb = base + (size_t)(tj * 16 + la) * DR;   // B col n -> fv row n
    const int kaoff = lh ? 8 : 0;   // A: lanes 16-31 hold K+8
    const int kboff = lh ? 8 : 0;   // B: lanes 16-31 hold K+8

    v8f acc = zero8();
    for (int kk = 0; kk < DR; kk += 32) {
        bf16x8 a0 = *(const bf16x8*)(pa + kk + kaoff);
        bf16x8 a1 = *(const bf16x8*)(pa + kk + kaoff + 16);
        bf16x8 b0 = *(const bf16x8*)(pb + kk + kboff);
        bf16x8 b1 = *(const bf16x8*)(pb + kk + kboff + 16);
        v16bf av, bv;
#pragma unroll
        for (int i = 0; i < 8; ++i) {
            av[i] = a0[i]; av[i + 8] = a1[i];
            bv[i] = b0[i]; bv[i + 8] = b1[i];
        }
        acc = __builtin_amdgcn_wmma_f32_16x16x32_bf16(
            false, av, false, bv, (short)0, acc, false, false);
        __builtin_prefetch(pa + kk + 256, 0, 1);
        __builtin_prefetch(pb + kk + 256, 0, 1);
    }

    // adjacency epilogue: A = ((d2 < DTH^2) | (cos > CTH)) + I
    const float* ctrb = ctr + (size_t)b * NN * 2;
    const float* rnb  = rn  + b * NN;
    const int n   = tj * 16 + la;
    const float rnn = rnb[n];
    const float cnx = ctrb[2 * n], cny = ctrb[2 * n + 1];
    float* Ab = Araw + (size_t)b * NN * NN;
#pragma unroll
    for (int r = 0; r < 8; ++r) {
        const int m = ti * 16 + r + 8 * lh;
        const float cosv = acc[r] * rnb[m] * rnn;
        const float dx = ctrb[2 * m] - cnx;
        const float dy = ctrb[2 * m + 1] - cny;
        const float d2 = dx * dx + dy * dy;
        float a = ((d2 < DTH2) || (cosv > CTH)) ? 1.0f : 0.0f;
        if (m == n) a += 1.0f;            // GCN self-loop
        Ab[m * NN + n] = a;
    }
}

// =====================================================================
// Kernel 2: symmetric degree normalization D^-1/2 A D^-1/2 (in place)
// =====================================================================
__global__ void norm_adj_kernel(float* __restrict__ A) {
    const int b = blockIdx.x;
    const int m = threadIdx.x;      // 128 threads
    float* Ab = A + (size_t)b * NN * NN;
    __shared__ float dinv[NN];
    float s = 0.0f;
    for (int n = 0; n < NN; ++n) s += Ab[m * NN + n];
    dinv[m] = rsqrtf(s);
    __syncthreads();
    const float dm = dinv[m];
    for (int n = 0; n < NN; ++n) Ab[m * NN + n] *= dm * dinv[n];
}

// =====================================================================
// Kernel 3: xw1 = [fv|ctr] @ W1   (bf16 WMMA over K=12544, + f32 rank-2)
// one wave per tile: 128 row tiles x 8 col tiles = 1024 waves
// =====================================================================
__global__ void xw1_kernel(const __bf16* __restrict__ fvb,
                           const __bf16* __restrict__ w1t,
                           const float*  __restrict__ ctr,
                           const float*  __restrict__ W1f,
                           float* __restrict__ out) {
    const int wave = (blockIdx.x * blockDim.x + threadIdx.x) >> 5;
    const int lane = threadIdx.x & 31;
    const int ti = wave >> 3, tj = wave & 7;
    const int la = lane & 15, lh = lane >> 4;

    const __bf16* pa = fvb + (size_t)(ti * 16 + la) * DR;    // x row
    const __bf16* pb = w1t + (size_t)(tj * 16 + la) * DR;    // W1 column h
    const int kaoff = lh ? 8 : 0;
    const int kboff = lh ? 8 : 0;

    v8f acc = zero8();
    for (int kk = 0; kk < DR; kk += 32) {
        bf16x8 a0 = *(const bf16x8*)(pa + kk + kaoff);
        bf16x8 a1 = *(const bf16x8*)(pa + kk + kaoff + 16);
        bf16x8 b0 = *(const bf16x8*)(pb + kk + kboff);
        bf16x8 b1 = *(const bf16x8*)(pb + kk + kboff + 16);
        v16bf av, bv;
#pragma unroll
        for (int i = 0; i < 8; ++i) {
            av[i] = a0[i]; av[i + 8] = a1[i];
            bv[i] = b0[i]; bv[i + 8] = b1[i];
        }
        acc = __builtin_amdgcn_wmma_f32_16x16x32_bf16(
            false, av, false, bv, (short)0, acc, false, false);
        __builtin_prefetch(pa + kk + 256, 0, 1);
        __builtin_prefetch(pb + kk + 256, 0, 1);
    }

    const int h = tj * 16 + la;
    const float w1x = W1f[(size_t)DR * HID + h];        // row 12544 (ctr.x)
    const float w1y = W1f[(size_t)(DR + 1) * HID + h];  // row 12545 (ctr.y)
#pragma unroll
    for (int r = 0; r < 8; ++r) {
        const int row = ti * 16 + r + 8 * lh;           // global row in [0,2048)
        const float cx = ctr[row * 2], cy = ctr[row * 2 + 1];
        out[(size_t)row * HID + h] = acc[r] + cx * w1x + cy * w1y;
    }
}

// =====================================================================
// Kernel 4: generic per-image 128x128 f32 GEMM via v_wmma_f32_16x16x4_f32
// Y[b] = act(A[b?]@X[b?] + bias); strides select per-image vs shared operand
// =====================================================================
__global__ void gemm128_kernel(const float* __restrict__ A, long strideA,
                               const float* __restrict__ X, long strideX,
                               const float* __restrict__ bias, int doRelu,
                               float* __restrict__ Y) {
    const int wave = (blockIdx.x * blockDim.x + threadIdx.x) >> 5;
    const int lane = threadIdx.x & 31;
    const int b  = wave >> 6;
    const int t  = wave & 63;
    const int ti = t >> 3, tj = t & 7;
    const int la = lane & 15, lh = lane >> 4;

    const float* Ab = A + (size_t)b * strideA;
    const float* Xb = X + (size_t)b * strideX;
    const int m = ti * 16 + la;     // A row for this lane
    const int h = tj * 16 + la;     // B column for this lane

    v8f acc = zero8();
    for (int kk = 0; kk < NN; kk += 4) {
        const int k0 = kk + 2 * lh; // lanes 16-31 carry K+2
        v2f a, bb;
        a.x  = Ab[m * NN + k0];
        a.y  = Ab[m * NN + k0 + 1];
        bb.x = Xb[k0 * HID + h];
        bb.y = Xb[(k0 + 1) * HID + h];
        acc = __builtin_amdgcn_wmma_f32_16x16x4_f32(
            false, a, false, bb, (short)0, acc, false, false);
    }
#pragma unroll
    for (int r = 0; r < 8; ++r) {
        const int mr = ti * 16 + r + 8 * lh;
        float v = acc[r];
        if (bias)   v += bias[h];
        if (doRelu) v = fmaxf(v, 0.0f);
        Y[(size_t)b * NN * HID + mr * HID + h] = v;
    }
}

// =====================================================================
// Kernel 5: global mean pool over N + 128->22 classifier
// =====================================================================
__global__ void pool_classify_kernel(const float* __restrict__ h2,
                                     const float* __restrict__ Wc,
                                     const float* __restrict__ bc,
                                     float* __restrict__ out) {
    const int b = blockIdx.x;
    const int h = threadIdx.x;      // 128 threads
    __shared__ float hg[HID];
    const float* Hb = h2 + (size_t)b * NN * HID;
    float s = 0.0f;
    for (int n = 0; n < NN; ++n) s += Hb[n * HID + h];
    hg[h] = s * (1.0f / (float)NN);
    __syncthreads();
    if (h < PARTS) {
        float acc = bc[h];
        for (int k = 0; k < HID; ++k) acc += hg[k] * Wc[k * PARTS + h];
        out[b * PARTS + h] = acc;
    }
}

// =====================================================================
// launcher
// =====================================================================
extern "C" void kernel_launch(void* const* d_in, const int* in_sizes, int n_in,
                              void* d_out, int out_size, void* d_ws, size_t ws_size,
                              hipStream_t stream) {
    (void)in_sizes; (void)n_in; (void)out_size; (void)ws_size;
    const float* fv  = (const float*)d_in[0];
    const float* ctr = (const float*)d_in[1];
    const float* W1  = (const float*)d_in[2];
    const float* b1  = (const float*)d_in[3];
    const float* W2  = (const float*)d_in[4];
    const float* b2  = (const float*)d_in[5];
    const float* Wc  = (const float*)d_in[6];
    const float* bc  = (const float*)d_in[7];
    float* out = (float*)d_out;

    char* ws = (char*)d_ws;
    const size_t szFVB = (size_t)2 * ROWS * DR;        // 51,380,224 B (bf16 fv)
    const size_t szW1T = (size_t)2 * HID * DR;         //  3,211,264 B (bf16 W1^T)
    const size_t szRN  = (size_t)4 * ROWS;             //      8,192 B
    const size_t szMAT = (size_t)4 * BI * NN * HID;    //  1,048,576 B each

    __bf16* fvb  = (__bf16*)(ws);
    __bf16* w1t  = (__bf16*)(ws + szFVB);
    float*  rn   = (float*)(ws + szFVB + szW1T);
    float*  ahat = (float*)(ws + szFVB + szW1T + szRN);
    float*  xw1  = (float*)((char*)ahat + szMAT);
    float*  h1   = (float*)((char*)xw1 + szMAT);
    float*  tbuf = (float*)((char*)h1  + szMAT);
    float*  h2   = (float*)((char*)tbuf + szMAT);

    // pass 0: precision conversion + norms
    convert_fv_kernel<<<ROWS, 256, 0, stream>>>(fv, fvb, rn);
    convert_w1t_kernel<<<1024, 256, 0, stream>>>(W1, w1t);

    // adjacency: bf16 Gram + thresholds, then D^-1/2 A D^-1/2
    gram_adj_kernel<<<128, 256, 0, stream>>>(fvb, rn, ctr, ahat);   // 1024 waves
    norm_adj_kernel<<<BI, NN, 0, stream>>>(ahat);

    // x @ W1 (bf16 WMMA K=12544 + f32 ctr rank-2)
    xw1_kernel<<<128, 256, 0, stream>>>(fvb, w1t, ctr, W1, xw1);

    // GCN layer 1: h1 = relu(A_hat @ xw1 + b1)
    gemm128_kernel<<<128, 256, 0, stream>>>(ahat, (long)(NN * NN),
                                            xw1, (long)(NN * HID), b1, 1, h1);
    // t = h1 @ W2   (W2 shared across images -> stride 0)
    gemm128_kernel<<<128, 256, 0, stream>>>(h1, (long)(NN * HID),
                                            W2, 0L, nullptr, 0, tbuf);
    // GCN layer 2: h2 = relu(A_hat @ t + b2)
    gemm128_kernel<<<128, 256, 0, stream>>>(ahat, (long)(NN * NN),
                                            tbuf, (long)(NN * HID), b2, 1, h2);

    // mean pool + classifier
    pool_classify_kernel<<<BI, HID, 0, stream>>>(h2, Wc, bc, out);
}
